// CausalSelfAttention_22084721836214
// MI455X (gfx1250) — matrix operand order repaired and checked by reading, not compiled, not run on hardware
//
#include <hip/hip_runtime.h>
#include <hip/hip_bf16.h>
#include <stdint.h>

#define B_   32
#define T_   512
#define C_   768
#define H_   12
#define D_   64
#define M_   (B_ * T_)     // 16384
#define NQKV (3 * C_)      // 2304

typedef __attribute__((ext_vector_type(16))) _Float16 v16h;
typedef __attribute__((ext_vector_type(8)))  _Float16 v8h;
typedef __attribute__((ext_vector_type(4)))  _Float16 v4h;
typedef __attribute__((ext_vector_type(8)))  float    v8f;
typedef __attribute__((ext_vector_type(4)))  uint32_t u32x4;
typedef __attribute__((ext_vector_type(8)))  int      i32x8;
typedef __attribute__((ext_vector_type(4)))  int      i32x4;

__device__ __forceinline__ v8f wmma_16x16x32(v16h a, v16h b, v8f c) {
  // (neg_a, A, neg_b, B, c_mod, C, reuse_a, reuse_b)
  return __builtin_amdgcn_wmma_f32_16x16x32_f16(false, a, false, b, (short)0, c,
                                                false, false);
}

// Build a 16-half fragment from two 16-byte LDS chunks (two ds_load_b128).
__device__ __forceinline__ v16h ldfrag(const _Float16* p0, const _Float16* p1) {
  v8h lo = *(const v8h*)p0;
  v8h hi = *(const v8h*)p1;
  return __builtin_shufflevector(lo, hi, 0, 1, 2, 3, 4, 5, 6, 7, 8, 9, 10, 11,
                                 12, 13, 14, 15);
}

// ---------------------------------------------------------------------------
// Tensor Data Mover: 2-D f16 tile (tile_w x tile_h elems, row stride in elems)
// global -> LDS, with LDS row padding (pad_amount DWORDs after every
// 2^(pad_int_code+1) DWORDs). D# per CDNA5 ISA.
// Issue once per workgroup (wave 0); EXEC is ignored by TDM.
// ---------------------------------------------------------------------------
__device__ __forceinline__ void tdm_load_2d_f16(
    const _Float16* gptr, uint32_t lds_off, uint32_t tile_w, uint32_t tile_h,
    uint32_t row_stride, uint32_t pad_int_code, uint32_t pad_amt_code) {
  const uint64_t ga = (uint64_t)(uintptr_t)gptr;
  u32x4 g0;
  g0[0] = 1u;                                             // count=1 (valid)
  g0[1] = lds_off;                                        // lds_addr (bytes)
  g0[2] = (uint32_t)ga;                                   // global_addr[31:0]
  g0[3] = (uint32_t)((ga >> 32) & 0x01FFFFFFu) | (2u << 30);  // [56:32]|type=2

  i32x8 g1;
  g1[0] = (int)((1u << 16) |                 // data_size = 1 (2 bytes)
                (1u << 20) |                 // pad_enable
                (pad_int_code << 22) |       // pad_interval
                (pad_amt_code << 25));       // pad_amount
  g1[1] = (int)((row_stride & 0xFFFFu) << 16);        // tensor_dim0[15:0]
  g1[2] = (int)((row_stride >> 16) | ((tile_h & 0xFFFFu) << 16));
  g1[3] = (int)((tile_h >> 16) | (tile_w << 16));     // tile_dim0
  g1[4] = (int)tile_h;                                // tile_dim1 (dim2 = 0)
  g1[5] = (int)row_stride;                            // tensor_dim0_stride
  g1[6] = 0;
  g1[7] = 0;

  const i32x4 z4 = {0, 0, 0, 0};
#if __clang_major__ >= 23
  const i32x8 z8 = {0, 0, 0, 0, 0, 0, 0, 0};
  __builtin_amdgcn_tensor_load_to_lds(g0, g1, z4, z4, z8, 0);
#else
  __builtin_amdgcn_tensor_load_to_lds(g0, g1, z4, z4, 0);
#endif
}

__device__ __forceinline__ uint32_t lds_offset(const void* p) {
  return (uint32_t)(uintptr_t)p;  // low 32 bits of flat shared addr = LDS byte addr
}

// ---------------------------------------------------------------------------
// Kernel 1: qkv = x @ w_qkv + b_qkv, output scattered as f16 [3][B][H][T][D]
// Block: 128 threads (4 waves). Tile: 128(M) x 64(N), K-step 32.
// (Inputs are f32 -> conversion needed, so staging stays VALU/VMEM here.)
// ---------------------------------------------------------------------------
__global__ __launch_bounds__(128) void qkv_gemm_kernel(
    const float* __restrict__ x, const float* __restrict__ wq,
    const float* __restrict__ bq, _Float16* __restrict__ qkv) {
  __shared__ __align__(16) _Float16 As[128][40];  // [m][k], padded row (80B)
  __shared__ __align__(16) _Float16 Bs[64][40];   // [n][k] (= B^T), padded

  const int tid = threadIdx.x;
  const int w = tid >> 5, lane = tid & 31, hi = lane >> 4, ln = lane & 15;
  const int n0 = blockIdx.x * 64;
  const int m0 = blockIdx.y * 128;

  v8f acc[2][4];
#pragma unroll
  for (int i = 0; i < 2; i++)
#pragma unroll
    for (int j = 0; j < 4; j++) acc[i][j] = (v8f){0, 0, 0, 0, 0, 0, 0, 0};

  for (int kc = 0; kc < C_; kc += 32) {
    // Stage A tile (f32 -> f16): float4 loads, v4h stores.
    for (int idx = tid; idx < 128 * 8; idx += 128) {
      int r = idx >> 3, q = (idx & 7) * 4;
      const float4 f = *(const float4*)&x[(size_t)(m0 + r) * C_ + kc + q];
      v4h h4;
      h4[0] = (_Float16)f.x; h4[1] = (_Float16)f.y;
      h4[2] = (_Float16)f.z; h4[3] = (_Float16)f.w;
      *(v4h*)&As[r][q] = h4;
    }
    // Stage B^T tile: w_qkv is [K][N]; float4 along N, transposed b16 stores.
    for (int idx = tid; idx < 32 * 16; idx += 128) {
      int kk = idx >> 4, n4 = (idx & 15) * 4;
      const float4 f = *(const float4*)&wq[(size_t)(kc + kk) * NQKV + n0 + n4];
      Bs[n4 + 0][kk] = (_Float16)f.x;
      Bs[n4 + 1][kk] = (_Float16)f.y;
      Bs[n4 + 2][kk] = (_Float16)f.z;
      Bs[n4 + 3][kk] = (_Float16)f.w;
    }
    if (kc + 32 < C_)
      __builtin_prefetch(&x[(size_t)(m0 + (tid >> 3)) * C_ + kc + 32], 0, 1);
    __syncthreads();

    v16h afrag[2], bfrag[4];
#pragma unroll
    for (int mi = 0; mi < 2; mi++) {
      const _Float16* rowp = &As[w * 32 + mi * 16 + ln][0];
      afrag[mi] = ldfrag(rowp + 8 * hi, rowp + 16 + 8 * hi);
    }
#pragma unroll
    for (int nj = 0; nj < 4; nj++) {
      const _Float16* rowp = &Bs[nj * 16 + ln][0];
      bfrag[nj] = ldfrag(rowp + 8 * hi, rowp + 16 + 8 * hi);
    }
#pragma unroll
    for (int mi = 0; mi < 2; mi++)
#pragma unroll
      for (int nj = 0; nj < 4; nj++)
        acc[mi][nj] = wmma_16x16x32(afrag[mi], bfrag[nj], acc[mi][nj]);
    __syncthreads();
  }

  // Epilogue: + bias, scatter f16 into [which][b][h][t][d].
#pragma unroll
  for (int mi = 0; mi < 2; mi++) {
#pragma unroll
    for (int nj = 0; nj < 4; nj++) {
      const int col = n0 + nj * 16 + ln;
      const float bias = bq[col];
      const int which = col / C_;
      const int rem = col - which * C_;
      const int h = rem >> 6, d = rem & 63;
#pragma unroll
      for (int r = 0; r < 8; r++) {
        const int row = m0 + w * 32 + mi * 16 + r + 8 * hi;
        const int b = row >> 9, t = row & (T_ - 1);
        const float v = acc[mi][nj][r] + bias;
        qkv[((((size_t)which * B_ + b) * H_ + h) * T_ + t) * D_ + d] =
            (_Float16)v;
      }
    }
  }
}

// ---------------------------------------------------------------------------
// Kernel 2: flash attention, one block per (b,h,qtile of 64). 4 waves x 16 q.
// Q and K tiles are DMA'd global->LDS by the Tensor Data Mover (with LDS row
// padding applied by the TDM); V^T needs a transpose so it is staged manually.
// ---------------------------------------------------------------------------
__global__ __launch_bounds__(128) void attn_kernel(
    const _Float16* __restrict__ qkv, _Float16* __restrict__ attn) {
  __shared__ __align__(16) _Float16 Qs[64][72];     // [q][d]   (TDM, pad 8)
  __shared__ __align__(16) _Float16 Ks[32][72];     // [key][d] (TDM, pad 8)
  __shared__ __align__(16) _Float16 Vts[64][40];    // [d][key] (= V^T)
  __shared__ __align__(16) _Float16 Ps[4][16][40];  // per-wave P [q][key]

  const int tid = threadIdx.x;
  const int w = tid >> 5, lane = tid & 31, hi = lane >> 4, ln = lane & 15;
  const int bh = blockIdx.x >> 3;  // 0..B*H-1
  const int qb = blockIdx.x & 7;
  const int q0 = qb * 64;
  const size_t NBH = (size_t)B_ * H_;
  const _Float16* qp = qkv + (size_t)bh * T_ * D_;
  const _Float16* kp = qkv + (NBH + bh) * (size_t)T_ * D_;
  const _Float16* vp = qkv + (2 * NBH + bh) * (size_t)T_ * D_;

  // Stage Q tile via TDM: 64x64 f16, row 128B -> pad_interval code 4 (16x8B),
  // pad 4 DWORDs (code 3) => 72-half padded rows.
  if (w == 0) {
    tdm_load_2d_f16(qp + (size_t)q0 * D_, lds_offset(&Qs[0][0]),
                    /*tile_w=*/64, /*tile_h=*/64, /*stride=*/64,
                    /*pad_int=*/4, /*pad_amt=*/3);
    __builtin_amdgcn_s_wait_tensorcnt(0);
  }
  __syncthreads();

  // Per-wave Q A-fragments, kept in registers for the whole key loop.
  v16h aQ[2];
  {
    const _Float16* rowp = &Qs[w * 16 + ln][0];
    aQ[0] = ldfrag(rowp + 8 * hi, rowp + 16 + 8 * hi);       // d = 0..31
    aQ[1] = ldfrag(rowp + 32 + 8 * hi, rowp + 48 + 8 * hi);  // d = 32..63
  }

  float m_run[8], l_run[8];
  v8f accO[4];
#pragma unroll
  for (int r = 0; r < 8; r++) { m_run[r] = -1e30f; l_run[r] = 0.f; }
#pragma unroll
  for (int i = 0; i < 4; i++) accO[i] = (v8f){0, 0, 0, 0, 0, 0, 0, 0};

  const int nkt = (q0 + 64) >> 5;      // causal bound for this block
  const int qw_hi = q0 + w * 16 + 15;  // this wave's last query row

  for (int kt = 0; kt < nkt; ++kt) {
    const int kbase = kt * 32;
    // K tile 32x64 via TDM (same pad config as Q).
    if (w == 0)
      tdm_load_2d_f16(kp + (size_t)kbase * D_, lds_offset(&Ks[0][0]),
                      /*tile_w=*/64, /*tile_h=*/32, /*stride=*/64,
                      /*pad_int=*/4, /*pad_amt=*/3);
    // V^T tile (transpose scatter) staged by all threads.
    for (int idx = tid; idx < 32 * 64; idx += 128) {
      int r = idx >> 6, d = idx & 63;
      Vts[d][r] = vp[(size_t)(kbase + r) * D_ + d];
    }
    if (w == 0) __builtin_amdgcn_s_wait_tensorcnt(0);
    __syncthreads();

    if (kbase <= qw_hi) {
      // S^tile = Q K^T : two 16-key sub-tiles, 2 WMMAs each (d = 2 x 32).
      v8f s[2];
#pragma unroll
      for (int nt = 0; nt < 2; nt++) {
        const _Float16* rowp = &Ks[nt * 16 + ln][0];
        v16h b0 = ldfrag(rowp + 8 * hi, rowp + 16 + 8 * hi);
        v16h b1 = ldfrag(rowp + 32 + 8 * hi, rowp + 48 + 8 * hi);
        v8f z = (v8f){0, 0, 0, 0, 0, 0, 0, 0};
        z = wmma_16x16x32(aQ[0], b0, z);
        z = wmma_16x16x32(aQ[1], b1, z);
        s[nt] = z;
      }
      // Scale 1/sqrt(D) and causal mask.
#pragma unroll
      for (int nt = 0; nt < 2; nt++) {
        const int key = kbase + nt * 16 + ln;
#pragma unroll
        for (int r = 0; r < 8; r++) {
          const int qrow = q0 + w * 16 + r + 8 * hi;
          const float v = s[nt][r] * 0.125f;
          s[nt][r] = (key <= qrow) ? v : -1e30f;
        }
      }
      // Online softmax: row stats reduce across the 16-lane half.
#pragma unroll
      for (int r = 0; r < 8; r++) {
        float mx = fmaxf(s[0][r], s[1][r]);
#pragma unroll
        for (int off = 8; off >= 1; off >>= 1)
          mx = fmaxf(mx, __shfl_xor(mx, off, 32));
        const float m_new = fmaxf(m_run[r], mx);
        const float alpha = __expf(m_run[r] - m_new);
        m_run[r] = m_new;
        const float p0 = __expf(s[0][r] - m_new);
        const float p1 = __expf(s[1][r] - m_new);
        float sum = p0 + p1;
#pragma unroll
        for (int off = 8; off >= 1; off >>= 1)
          sum += __shfl_xor(sum, off, 32);
        l_run[r] = l_run[r] * alpha + sum;
#pragma unroll
        for (int i = 0; i < 4; i++) accO[i][r] *= alpha;
        Ps[w][8 * hi + r][ln] = (_Float16)p0;
        Ps[w][8 * hi + r][16 + ln] = (_Float16)p1;
      }
      // O += P V : A = P (16x32 keys), B^T = V^T rows (d-major).
      {
        const _Float16* prow = &Ps[w][ln][0];
        v16h aP = ldfrag(prow + 8 * hi, prow + 16 + 8 * hi);
#pragma unroll
        for (int i = 0; i < 4; i++) {
          const _Float16* vrow = &Vts[i * 16 + ln][0];
          v16h bV = ldfrag(vrow + 8 * hi, vrow + 16 + 8 * hi);
          accO[i] = wmma_16x16x32(aP, bV, accO[i]);
        }
      }
    }
    __syncthreads();
  }

  // Epilogue: attn[(b*T + q)*C + h*64 + d] as f16 (feeds proj GEMM directly).
  const int b = bh / H_, h = bh - b * H_;
#pragma unroll
  for (int i = 0; i < 4; i++) {
    const int d = i * 16 + ln;
#pragma unroll
    for (int r = 0; r < 8; r++) {
      const int qrow = q0 + w * 16 + r + 8 * hi;
      const float o = accO[i][r] / l_run[r];
      attn[(size_t)(b * T_ + qrow) * C_ + h * D_ + d] = (_Float16)o;
    }
  }
}

// ---------------------------------------------------------------------------
// Kernel 3: out = attn @ w_proj + b_proj (fp32 out).
// A tile (f16, strided 2-D) is DMA'd by the TDM; B needs f32->f16 transpose.
// ---------------------------------------------------------------------------
__global__ __launch_bounds__(128) void proj_gemm_kernel(
    const _Float16* __restrict__ a, const float* __restrict__ wp,
    const float* __restrict__ bp, float* __restrict__ out) {
  __shared__ __align__(16) _Float16 As[128][40];  // TDM, row 64B + 16B pad
  __shared__ __align__(16) _Float16 Bs[64][40];

  const int tid = threadIdx.x;
  const int w = tid >> 5, lane = tid & 31, hi = lane >> 4, ln = lane & 15;
  const int n0 = blockIdx.x * 64;
  const int m0 = blockIdx.y * 128;

  v8f acc[2][4];
#pragma unroll
  for (int i = 0; i < 2; i++)
#pragma unroll
    for (int j = 0; j < 4; j++) acc[i][j] = (v8f){0, 0, 0, 0, 0, 0, 0, 0};

  for (int kc = 0; kc < C_; kc += 32) {
    // A tile via TDM: 32x128 f16 out of 768-stride tensor; row 64B ->
    // pad_interval code 3 (8x8B), pad 4 DWORDs (code 3) => 40-half rows.
    if (w == 0)
      tdm_load_2d_f16(a + (size_t)m0 * C_ + kc, lds_offset(&As[0][0]),
                      /*tile_w=*/32, /*tile_h=*/128, /*stride=*/C_,
                      /*pad_int=*/3, /*pad_amt=*/3);
    // Stage B^T tile (f32 -> f16 transpose).
    for (int idx = tid; idx < 32 * 16; idx += 128) {
      int kk = idx >> 4, n4 = (idx & 15) * 4;
      const float4 f = *(const float4*)&wp[(size_t)(kc + kk) * C_ + n0 + n4];
      Bs[n4 + 0][kk] = (_Float16)f.x;
      Bs[n4 + 1][kk] = (_Float16)f.y;
      Bs[n4 + 2][kk] = (_Float16)f.z;
      Bs[n4 + 3][kk] = (_Float16)f.w;
    }
    if (w == 0) __builtin_amdgcn_s_wait_tensorcnt(0);
    __syncthreads();

    v16h afrag[2], bfrag[4];
#pragma unroll
    for (int mi = 0; mi < 2; mi++) {
      const _Float16* rowp = &As[w * 32 + mi * 16 + ln][0];
      afrag[mi] = ldfrag(rowp + 8 * hi, rowp + 16 + 8 * hi);
    }
#pragma unroll
    for (int nj = 0; nj < 4; nj++) {
      const _Float16* rowp = &Bs[nj * 16 + ln][0];
      bfrag[nj] = ldfrag(rowp + 8 * hi, rowp + 16 + 8 * hi);
    }
#pragma unroll
    for (int mi = 0; mi < 2; mi++)
#pragma unroll
      for (int nj = 0; nj < 4; nj++)
        acc[mi][nj] = wmma_16x16x32(afrag[mi], bfrag[nj], acc[mi][nj]);
    __syncthreads();
  }

#pragma unroll
  for (int mi = 0; mi < 2; mi++) {
#pragma unroll
    for (int nj = 0; nj < 4; nj++) {
      const int col = n0 + nj * 16 + ln;
      const float bias = bp[col];
#pragma unroll
      for (int r = 0; r < 8; r++) {
        const int row = m0 + w * 32 + mi * 16 + r + 8 * hi;
        out[(size_t)row * C_ + col] = acc[mi][nj][r] + bias;
      }
    }
  }
}

// ---------------------------------------------------------------------------
extern "C" void kernel_launch(void* const* d_in, const int* in_sizes, int n_in,
                              void* d_out, int out_size, void* d_ws,
                              size_t ws_size, hipStream_t stream) {
  (void)in_sizes; (void)n_in; (void)out_size; (void)ws_size;
  const float* x  = (const float*)d_in[0];
  const float* wq = (const float*)d_in[1];
  const float* bq = (const float*)d_in[2];
  const float* wp = (const float*)d_in[3];
  const float* bp = (const float*)d_in[4];

  _Float16* qkv_ws  = (_Float16*)d_ws;                        // [3][B][H][T][D]
  _Float16* attn_ws = qkv_ws + (size_t)3 * B_ * H_ * T_ * D_;  // [M][C]

  qkv_gemm_kernel<<<dim3(NQKV / 64, M_ / 128), 128, 0, stream>>>(x, wq, bq,
                                                                 qkv_ws);
  attn_kernel<<<dim3(B_ * H_ * (T_ / 64)), 128, 0, stream>>>(qkv_ws, attn_ws);
  proj_gemm_kernel<<<dim3(C_ / 64, M_ / 128), 128, 0, stream>>>(attn_ws, wp,
                                                                bp,
                                                                (float*)d_out);
}
